// TimeAwareNodeModel_50440095924468
// MI455X (gfx1250) — compile-verified, matrix operand order checked by reading, not run on hardware
//
#include <hip/hip_runtime.h>
#include <hip/hip_bf16.h>

typedef __attribute__((ext_vector_type(16))) _Float16 v16h;
typedef __attribute__((ext_vector_type(8)))  _Float16 v8h;
typedef __attribute__((ext_vector_type(8)))  float    v8f;

#define N_    50000
#define E_    300000
#define D_    128
#define DE_   64
#define H_    4
#define HID_  128
#define OUT_  32
#define FINP_ 192     // D + DE
#define FLOW_ 256     // 2*H*OUT
#define NTILE_ (E_ / 16)   // 18750 edge tiles
#define NEGMAX (-3.4028234663852886e38f)

// ---------- helpers ----------
static __device__ __forceinline__ v16h hcombine(v8h lo, v8h hi) {
  v16h r;
#pragma unroll
  for (int i = 0; i < 8; ++i) { r[i] = lo[i]; r[i + 8] = hi[i]; }
  return r;
}
// A/B fragment from f16 memory: two contiguous 16-byte runs of 8 halves each
static __device__ __forceinline__ v16h load_frag(const _Float16* p0, const _Float16* p1) {
  return hcombine(*(const v8h*)p0, *(const v8h*)p1);
}
// 8 consecutive f32 -> 8 f16 (for building A fragments straight from global f32 data)
static __device__ __forceinline__ v8h cvt8(const float* p) {
  float4 a = *(const float4*)p;
  float4 b = *(const float4*)(p + 4);
  v8h r;
  r[0] = (_Float16)a.x; r[1] = (_Float16)a.y; r[2] = (_Float16)a.z; r[3] = (_Float16)a.w;
  r[4] = (_Float16)b.x; r[5] = (_Float16)b.y; r[6] = (_Float16)b.z; r[7] = (_Float16)b.w;
  return r;
}
// order-preserving float <-> unsigned for atomic segment-max
static __device__ __forceinline__ unsigned f2ord(float f) {
  unsigned u = __float_as_uint(f);
  return (u & 0x80000000u) ? ~u : (u | 0x80000000u);
}
static __device__ __forceinline__ float ord2f(unsigned u) {
  unsigned v = (u & 0x80000000u) ? (u & 0x7fffffffu) : ~u;
  return __uint_as_float(v);
}

// ---------- kernel 1: pack f32 weights into f16 WMMA B-fragment layout ----------
// B-frag element order per 512-half fragment: [lane 0..31][i 0..15]
// k = kt*32 + (i&7) + ((i>>3)<<4) + ((lane>>4)<<3), n = nt*16 + (lane&15)
__global__ void pack_weights_kernel(const float* __restrict__ W1o, const float* __restrict__ W1i,
                                    const float* __restrict__ W2o, const float* __restrict__ W2i,
                                    const float* __restrict__ Wn,
                                    _Float16* __restrict__ W1p, _Float16* __restrict__ W2p,
                                    _Float16* __restrict__ Wnp) {
  int idx = blockIdx.x * 256 + threadIdx.x;
  if (idx >= 262144) return;
  if (idx < 196608) {                       // W1: 2 dir * 4 h * (6 kt * 8 nt) frags
    int u = idx >> 9, w = idx & 511;
    int lane = w >> 4, i = w & 15;
    int frag = u % 48, dh = u / 48;
    int dir = dh >> 2, h = dh & 3;
    int kt = frag >> 3, nt = frag & 7;
    int n = nt * 16 + (lane & 15);
    int k = kt * 32 + (i & 7) + ((i >> 3) << 4) + ((lane >> 4) << 3);
    const float* W = dir ? W1i : W1o;       // [H,192,128]
    W1p[idx] = (_Float16)W[((size_t)h * FINP_ + k) * HID_ + n];
  } else if (idx < 229376) {                // W2: 2 dir * 4 h * (4 kt * 2 nt) frags
    int j = idx - 196608;
    int u = j >> 9, w = j & 511;
    int lane = w >> 4, i = w & 15;
    int frag = u & 7, dh = u >> 3;
    int dir = dh >> 2, h = dh & 3;
    int kt = frag >> 1, nt = frag & 1;
    int n = nt * 16 + (lane & 15);
    int k = kt * 32 + (i & 7) + ((i >> 3) << 4) + ((lane >> 4) << 3);
    const float* W = dir ? W2i : W2o;       // [H,128,32]
    W2p[j] = (_Float16)W[((size_t)h * HID_ + k) * OUT_ + n];
  } else {                                  // Wn: 8 kt * 8 nt frags, [256,128]
    int j = idx - 229376;
    int u = j >> 9, w = j & 511;
    int lane = w >> 4, i = w & 15;
    int kt = u >> 3, nt = u & 7;
    int n = nt * 16 + (lane & 15);
    int k = kt * 32 + (i & 7) + ((i >> 3) << 4) + ((lane >> 4) << 3);
    Wnp[j] = (_Float16)Wn[(size_t)k * D_ + n];
  }
}

// ---------- kernel 2: zero/neutral init of scratch ----------
__global__ void init_kernel(float* __restrict__ agg,
                            unsigned* __restrict__ smax_o, unsigned* __restrict__ smax_i,
                            float* __restrict__ den_o, float* __restrict__ den_i) {
  int i = blockIdx.x * 256 + threadIdx.x;
  if (i < N_ * FLOW_) agg[i] = 0.f;
  if (i < N_ * H_) { smax_o[i] = 0u; smax_i[i] = 0u; den_o[i] = 0.f; den_i[i] = 0.f; }
}

// ---------- kernel 3: attention scores + masked leaky-relu + atomic segment max ----------
__global__ __launch_bounds__(256) void att_scores_kernel(
    const float* __restrict__ x, const int* __restrict__ ei,
    const float* __restrict__ aa_out, const float* __restrict__ aa_in,
    float* __restrict__ att_o, float* __restrict__ att_i,
    unsigned* __restrict__ smax_o, unsigned* __restrict__ smax_i) {
  __shared__ float s_aa[2048];              // [out|in][h][256]
  int tid = threadIdx.x;
  for (int i = tid; i < 2048; i += 256) s_aa[i] = (i < 1024) ? aa_out[i] : aa_in[i - 1024];
  __syncthreads();
  int e = blockIdx.x * 256 + tid;
  if (e >= E_) return;
  int r = ei[e], c = ei[E_ + e];
  const float4* xr = (const float4*)(x + (size_t)r * D_);
  const float4* xc = (const float4*)(x + (size_t)c * D_);
  float ao[4] = {0.f, 0.f, 0.f, 0.f}, ai[4] = {0.f, 0.f, 0.f, 0.f};
  for (int d = 0; d < 32; ++d) {
    float4 a = xr[d], b = xc[d];
#pragma unroll
    for (int h = 0; h < 4; ++h) {
      const float* po = &s_aa[h * 256 + 4 * d];
      const float* pi = &s_aa[1024 + h * 256 + 4 * d];
      ao[h] += a.x * po[0] + a.y * po[1] + a.z * po[2] + a.w * po[3]
             + b.x * po[128] + b.y * po[129] + b.z * po[130] + b.w * po[131];
      ai[h] += a.x * pi[0] + a.y * pi[1] + a.z * pi[2] + a.w * pi[3]
             + b.x * pi[128] + b.y * pi[129] + b.z * pi[130] + b.w * pi[131];
    }
  }
  bool mo = r < c;
#pragma unroll
  for (int h = 0; h < 4; ++h) {
    float so = ao[h]; so = so > 0.f ? so : 0.2f * so;
    float si = ai[h]; si = si > 0.f ? si : 0.2f * si;
    so = mo ? so : NEGMAX;
    si = mo ? NEGMAX : si;
    att_o[(size_t)h * E_ + e] = so;
    att_i[(size_t)h * E_ + e] = si;
    atomicMax(&smax_o[r * H_ + h], f2ord(so));
    atomicMax(&smax_i[r * H_ + h], f2ord(si));
  }
}

// ---------- kernel 4: exp(e - smax) + atomic denominator ----------
__global__ __launch_bounds__(256) void exp_sum_kernel(
    const int* __restrict__ ei,
    float* __restrict__ att_o, float* __restrict__ att_i,
    const unsigned* __restrict__ smax_o, const unsigned* __restrict__ smax_i,
    float* __restrict__ den_o, float* __restrict__ den_i) {
  int e = blockIdx.x * 256 + threadIdx.x;
  if (e >= E_) return;
  int r = ei[e], c = ei[E_ + e];
  bool mo = r < c;
#pragma unroll
  for (int h = 0; h < 4; ++h) {
    float mxo = ord2f(smax_o[r * H_ + h]); if (!__builtin_isfinite(mxo)) mxo = 0.f;
    float mxi = ord2f(smax_i[r * H_ + h]); if (!__builtin_isfinite(mxi)) mxi = 0.f;
    float so = att_o[(size_t)h * E_ + e];
    float si = att_i[(size_t)h * E_ + e];
    float exo = mo ? __expf(so - mxo) : 0.f;
    float exi = mo ? 0.f : __expf(si - mxi);
    att_o[(size_t)h * E_ + e] = exo;
    att_i[(size_t)h * E_ + e] = exi;
    if (mo)  atomicAdd(&den_o[r * H_ + h], exo);
    if (!mo) atomicAdd(&den_i[r * H_ + h], exi);
  }
}

// ---------- kernel 5: normalize attention + write `a` output ----------
__global__ __launch_bounds__(256) void norm_kernel(
    const int* __restrict__ ei,
    float* __restrict__ att_o, float* __restrict__ att_i,
    const float* __restrict__ den_o, const float* __restrict__ den_i,
    float* __restrict__ a_dst) {
  int e = blockIdx.x * 256 + threadIdx.x;
  if (e >= E_) return;
  int r = ei[e], c = ei[E_ + e];
  bool mo = r < c;
#pragma unroll
  for (int h = 0; h < 4; ++h) {
    float ao = att_o[(size_t)h * E_ + e] / (den_o[r * H_ + h] + 1e-12f);
    float ai = att_i[(size_t)h * E_ + e] / (den_i[r * H_ + h] + 1e-12f);
    att_o[(size_t)h * E_ + e] = ao;
    att_i[(size_t)h * E_ + e] = ai;
    a_dst[(size_t)h * E_ + e] = mo ? ao : ai;
  }
}

// ---------- kernel 6: edge MLP via WMMA (dominant compute) ----------
// grid = (ceil(NTILE/4), H). block = 128 threads = 4 waves. All 4 waves share the
// SAME head (blockIdx.y) so their B-fragment streams hit the WGP$ together; each
// wave owns a distinct 16-edge tile. A fragments are built directly from global
// f32 x/edge_attr (the two 8-element K runs never straddle the x|edge_attr seam).
__global__ __launch_bounds__(128) void edge_mlp_kernel(
    const float* __restrict__ x, const float* __restrict__ ea, const int* __restrict__ ei,
    const _Float16* __restrict__ W1p, const _Float16* __restrict__ W2p,
    const float* __restrict__ b1_out, const float* __restrict__ b2_out,
    const float* __restrict__ b1_in,  const float* __restrict__ b2_in,
    const float* __restrict__ a_out,  const float* __restrict__ a_in,
    float* __restrict__ agg) {
  __shared__ _Float16 s_h1[4 * 16 * HID_];     // 16 KB: per-wave GEMM1->GEMM2 transpose
  const int tid = threadIdx.x;
  const int lane = tid & 31, wave = tid >> 5;
  const int tile = blockIdx.x * 4 + wave;
  if (tile >= NTILE_) return;
  const int h = blockIdx.y;
  const int ebase = tile * 16;
  const int m16 = lane & 15, hi = lane >> 4;

  int nodes[8];
#pragma unroll
  for (int rr = 0; rr < 8; ++rr) nodes[rr] = ei[ebase + rr + hi * 8];

  // A fragments for GEMM1, straight from global f32 (shared across both dirs)
  const float* xrow  = x  + (size_t)ei[ebase + m16] * D_;
  const float* earow = ea + (size_t)(ebase + m16) * DE_;
  v16h A1[6];
#pragma unroll
  for (int kt = 0; kt < 6; ++kt) {
    int k0 = kt * 32 + hi * 8;
    int k1 = k0 + 16;
    const float* p0 = (k0 < D_) ? (xrow + k0) : (earow + (k0 - D_));
    const float* p1 = (k1 < D_) ? (xrow + k1) : (earow + (k1 - D_));
    A1[kt] = hcombine(cvt8(p0), cvt8(p1));
  }
  _Float16* h1 = &s_h1[wave * 16 * HID_];

  for (int dir = 0; dir < 2; ++dir) {
    const _Float16* w1 = W1p + ((size_t)(dir * H_ + h) * 48) * 512;
    const _Float16* w2 = W2p + ((size_t)(dir * H_ + h) * 8) * 512;
    const float* b1  = (dir ? b1_in  : b1_out) + h * HID_;
    const float* b2  = (dir ? b2_in  : b2_out) + h * OUT_;
    const float* att = (dir ? a_in   : a_out)  + (size_t)h * E_;
    const int colbase = (dir == 0) ? (H_ * OUT_ + h * OUT_) : (h * OUT_);  // flow = [agg_in | agg_out]

    // GEMM1: [16x192] @ [192x128] -> relu -> f16 into LDS
#pragma unroll
    for (int nt = 0; nt < 8; ++nt) {
      v8f acc = {};
#pragma unroll
      for (int kt = 0; kt < 6; ++kt) {
        const _Float16* bp = w1 + (size_t)(kt * 8 + nt) * 512 + lane * 16;
        v16h B = load_frag(bp, bp + 8);
        acc = __builtin_amdgcn_wmma_f32_16x16x32_f16(false, A1[kt], false, B,
                                                     (short)0, acc, false, false);
      }
      float bias = b1[nt * 16 + m16];
#pragma unroll
      for (int rr = 0; rr < 8; ++rr) {
        float v = acc[rr] + bias;
        v = v > 0.f ? v : 0.f;
        h1[(rr + hi * 8) * HID_ + nt * 16 + m16] = (_Float16)v;
      }
    }
    asm volatile("s_wait_dscnt 0" ::: "memory");   // within-wave LDS produce->consume

    // GEMM2: [16x128] @ [128x32] -> +bias, * attention, atomic scatter-add
    v16h A2[4];
#pragma unroll
    for (int kt = 0; kt < 4; ++kt) {
      const _Float16* p = &h1[m16 * HID_ + kt * 32 + hi * 8];
      A2[kt] = load_frag(p, p + 16);
    }
    float avals[8];
#pragma unroll
    for (int rr = 0; rr < 8; ++rr) avals[rr] = att[ebase + rr + hi * 8];
#pragma unroll
    for (int nt = 0; nt < 2; ++nt) {
      v8f acc = {};
#pragma unroll
      for (int kt = 0; kt < 4; ++kt) {
        const _Float16* bp = w2 + (size_t)(kt * 2 + nt) * 512 + lane * 16;
        v16h B = load_frag(bp, bp + 8);
        acc = __builtin_amdgcn_wmma_f32_16x16x32_f16(false, A2[kt], false, B,
                                                     (short)0, acc, false, false);
      }
      int ncol = nt * 16 + m16;
      float bias2 = b2[ncol];
#pragma unroll
      for (int rr = 0; rr < 8; ++rr) {
        float msg = (acc[rr] + bias2) * avals[rr];
        atomicAdd(&agg[(size_t)nodes[rr] * FLOW_ + colbase + ncol], msg);
      }
    }
  }
}

// ---------- kernel 7: node MLP flow[N,256] @ Wn[256,128] + bn via WMMA ----------
// A fragments built directly from global f32 agg rows (contiguous), no LDS needed.
__global__ __launch_bounds__(128) void node_mlp_kernel(
    const float* __restrict__ agg, const _Float16* __restrict__ Wnp,
    const float* __restrict__ bn, float* __restrict__ out) {
  const int tid = threadIdx.x;
  const int lane = tid & 31, wave = tid >> 5;
  const int tile = blockIdx.x * 4 + wave;
  if (tile >= N_ / 16) return;
  const int nb = tile * 16;
  const int m16 = lane & 15, hi = lane >> 4;
  const float* rowp = agg + (size_t)(nb + m16) * FLOW_;
  v16h A[8];
#pragma unroll
  for (int kt = 0; kt < 8; ++kt) {
    int k0 = kt * 32 + hi * 8;
    A[kt] = hcombine(cvt8(rowp + k0), cvt8(rowp + k0 + 16));
  }
#pragma unroll
  for (int nt = 0; nt < 8; ++nt) {
    v8f acc = {};
#pragma unroll
    for (int kt = 0; kt < 8; ++kt) {
      const _Float16* bp = Wnp + (size_t)(kt * 8 + nt) * 512 + lane * 16;
      v16h B = load_frag(bp, bp + 8);
      acc = __builtin_amdgcn_wmma_f32_16x16x32_f16(false, A[kt], false, B,
                                                   (short)0, acc, false, false);
    }
    int ncol = nt * 16 + m16;
    float b = bn[ncol];
#pragma unroll
    for (int rr = 0; rr < 8; ++rr)
      out[(size_t)(nb + rr + hi * 8) * D_ + ncol] = acc[rr] + b;
  }
}

// ---------- launch ----------
extern "C" void kernel_launch(void* const* d_in, const int* in_sizes, int n_in,
                              void* d_out, int out_size, void* d_ws, size_t ws_size,
                              hipStream_t stream) {
  const float* x      = (const float*)d_in[0];
  const float* ea     = (const float*)d_in[1];
  const int*   ei     = (const int*)  d_in[2];   // [row | col]
  const float* W1o    = (const float*)d_in[3];
  const float* b1o    = (const float*)d_in[4];
  const float* W2o    = (const float*)d_in[5];
  const float* b2o    = (const float*)d_in[6];
  const float* W1i    = (const float*)d_in[7];
  const float* b1i    = (const float*)d_in[8];
  const float* W2i    = (const float*)d_in[9];
  const float* b2i    = (const float*)d_in[10];
  const float* aa_out = (const float*)d_in[11];
  const float* aa_in  = (const float*)d_in[12];
  const float* Wn     = (const float*)d_in[13];
  const float* bn     = (const float*)d_in[14];

  float* out   = (float*)d_out;                  // [N,128]
  float* a_dst = out + (size_t)N_ * D_;          // [H,E]

  // workspace layout
  char* ws = (char*)d_ws;
  _Float16* W1p = (_Float16*)ws;                 // 196608 halves
  _Float16* W2p = W1p + 196608;                  // 32768 halves
  _Float16* Wnp = W2p + 32768;                   // 32768 halves
  float*    att_o  = (float*)(ws + 524288);      // H*E
  float*    att_i  = att_o + (size_t)H_ * E_;
  unsigned* smax_o = (unsigned*)(att_i + (size_t)H_ * E_);  // N*H
  unsigned* smax_i = smax_o + (size_t)N_ * H_;
  float*    den_o  = (float*)(smax_i + (size_t)N_ * H_);
  float*    den_i  = den_o + (size_t)N_ * H_;
  float*    agg    = den_i + (size_t)N_ * H_;    // N*256

  pack_weights_kernel<<<1024, 256, 0, stream>>>(W1o, W1i, W2o, W2i, Wn, W1p, W2p, Wnp);
  init_kernel<<<(N_ * FLOW_ + 255) / 256, 256, 0, stream>>>(agg, smax_o, smax_i, den_o, den_i);
  att_scores_kernel<<<(E_ + 255) / 256, 256, 0, stream>>>(x, ei, aa_out, aa_in,
                                                          att_o, att_i, smax_o, smax_i);
  exp_sum_kernel<<<(E_ + 255) / 256, 256, 0, stream>>>(ei, att_o, att_i, smax_o, smax_i,
                                                       den_o, den_i);
  norm_kernel<<<(E_ + 255) / 256, 256, 0, stream>>>(ei, att_o, att_i, den_o, den_i, a_dst);
  edge_mlp_kernel<<<dim3((NTILE_ + 3) / 4, H_), 128, 0, stream>>>(
      x, ea, ei, W1p, W2p, b1o, b2o, b1i, b2i, att_o, att_i, agg);
  node_mlp_kernel<<<(N_ / 16 + 3) / 4, 128, 0, stream>>>(agg, Wnp, bn, out);
}
